// RNNDecoder_42073499632036
// MI455X (gfx1250) — matrix operand order repaired and checked by reading, not compiled, not run on hardware
//
#include <hip/hip_runtime.h>
#include <math.h>

// ---------------------------------------------------------------------------
// RNN decoder (LSTM + vocab projection + log_softmax) for gfx1250 / MI455X.
// bf16 WMMA (v_wmma_f32_16x16x32_bf16) for the LSTM gate GEMMs and the
// dominant [2048,512]x[512,32000] projection GEMM (LDS-tiled, 128x128 block
// per workgroup so L2 traffic drops from ~5 GB to ~1 GB). f32 accumulate.
// ---------------------------------------------------------------------------

typedef __bf16 bf16_t;
typedef __attribute__((ext_vector_type(16))) __bf16 v16bf;
typedef __attribute__((ext_vector_type(8)))  __bf16 v8bf;
typedef __attribute__((ext_vector_type(8)))  float  v8f;

constexpr int T_  = 64;            // target sequence length
constexpr int B_  = 32;            // batch
constexpr int H_  = 512;           // hidden size
constexpr int D_  = 512;           // embedding size
constexpr int V_  = 32000;         // vocab
constexpr int G_  = 4 * H_;        // 2048 gate width
constexpr int KC_ = D_ + H_;       // 1024 combined K for [x|h] GEMM

// --- WMMA fragment helpers (contiguous-K layout used by this file, wave32) ----------
// A 16x32 bf16: lane (m = lane&15, half = lane>>4) holds K in
//   [half*16, half*16+16) contiguous  -> two 16B loads.
__device__ inline v16bf load_a_frag(const bf16_t* __restrict__ row, int half) {
  const v8bf lo = *reinterpret_cast<const v8bf*>(row + half * 16);
  const v8bf hi = *reinterpret_cast<const v8bf*>(row + half * 16 + 8);
  v16bf a;
#pragma unroll
  for (int i = 0; i < 8; ++i) { a[i] = lo[i]; a[i + 8] = hi[i]; }
  return a;
}
// B 32x16 bf16: lane (n = lane&15, half = lane>>4) holds K in
//   [half*16, half*16+16) contiguous -> one 32B load from weight row n.
__device__ inline v16bf load_b_frag(const bf16_t* __restrict__ wrow, int kb, int half) {
  return *reinterpret_cast<const v16bf*>(wrow + kb + half * 16);
}

// --- prep kernels ----------------------------------------------------------
__global__ __launch_bounds__(256)
void k_pack_wcat(const float* __restrict__ W_ih, const float* __restrict__ W_hh,
                 bf16_t* __restrict__ Wcat) {          // [G_][KC_]
  int idx = blockIdx.x * 256 + threadIdx.x;            // G_*KC_ threads
  int n = idx >> 10, k = idx & 1023;
  float v = (k < D_) ? W_ih[(size_t)n * D_ + k] : W_hh[(size_t)n * H_ + (k - D_)];
  Wcat[idx] = (bf16_t)v;
}

__global__ __launch_bounds__(256)
void k_convert_wout(const float* __restrict__ W_out, bf16_t* __restrict__ Wo) {
  int idx = blockIdx.x * 256 + threadIdx.x;            // V_*H_ threads
  Wo[idx] = (bf16_t)W_out[idx];
}

__global__ __launch_bounds__(256)
void k_embed(const float* __restrict__ emb, const long long* __restrict__ dst,
             bf16_t* __restrict__ x_bf) {              // [T_*B_][D_]
  int idx = blockIdx.x * 256 + threadIdx.x;            // T_*B_*D_ threads
  int tb = idx >> 9, k = idx & 511;
  long long row = dst[tb];
  x_bf[idx] = (bf16_t)emb[(size_t)row * D_ + k];
}

__global__ __launch_bounds__(256)
void k_init_state(const float* __restrict__ h0, const float* __restrict__ c0,
                  bf16_t* __restrict__ h_bf, float* __restrict__ c_st) {
  int idx = blockIdx.x * 256 + threadIdx.x;            // B_*H_ threads
  h_bf[idx] = (bf16_t)h0[idx];
  c_st[idx] = c0[idx];
}

// --- LSTM step: gates = [x_t | h] @ Wcat^T + b_ih + b_hh -------------------
// M=32, N=2048, K=1024.  256 waves, one 16x16 tile each, 32 WMMAs over K.
__global__ __launch_bounds__(256)
void k_lstm_gates(const bf16_t* __restrict__ x_t,      // [B_][D_] (step slice)
                  const bf16_t* __restrict__ h_bf,     // [B_][H_]
                  const bf16_t* __restrict__ Wcat,     // [G_][KC_]
                  const float* __restrict__ b_ih, const float* __restrict__ b_hh,
                  float* __restrict__ gates) {         // [B_][G_]
  const int wave  = (blockIdx.x * blockDim.x + threadIdx.x) >> 5;
  const int lane  = threadIdx.x & 31;
  const int mtile = wave >> 7;      // 0..1
  const int ntile = wave & 127;     // 0..127
  const int m = lane & 15, half = lane >> 4;
  const int arow_i = mtile * 16 + m;
  const bf16_t* wrow = Wcat + (size_t)(ntile * 16 + m) * KC_;
  v8f acc = {};
#pragma unroll 4
  for (int kb = 0; kb < KC_; kb += 32) {
    // virtual concatenation of x_t and h along K (32-chunks never straddle 512)
    const bf16_t* arow = (kb < D_) ? (x_t + (size_t)arow_i * D_ + kb)
                                   : (h_bf + (size_t)arow_i * H_ + (kb - D_));
    v16bf a = load_a_frag(arow, half);
    v16bf b = load_b_frag(wrow, kb, half);
    acc = __builtin_amdgcn_wmma_f32_16x16x32_bf16(false, a, false, b,
                                                  (short)0, acc, false, false);
  }
  const int n = ntile * 16 + m;
  const float bias = b_ih[n] + b_hh[n];
#pragma unroll
  for (int r = 0; r < 8; ++r) {
    const int rm = mtile * 16 + half * 8 + r;          // D-layout: M = half*8 + r
    gates[(size_t)rm * G_ + n] = acc[r] + bias;
  }
}

__global__ __launch_bounds__(256)
void k_lstm_update(const float* __restrict__ gates, float* __restrict__ c_st,
                   bf16_t* __restrict__ h_bf, bf16_t* __restrict__ out_bf_t) {
  int idx = blockIdx.x * 256 + threadIdx.x;            // B_*H_ threads
  int b = idx >> 9, j = idx & 511;
  const float* g = gates + (size_t)b * G_;
  float ig = g[j], fg = g[H_ + j], gg = g[2 * H_ + j], og = g[3 * H_ + j];
  float i_s = 1.f / (1.f + __expf(-ig));
  float f_s = 1.f / (1.f + __expf(-fg));
  float o_s = 1.f / (1.f + __expf(-og));
  float c = f_s * c_st[idx] + i_s * tanhf(gg);
  float h = o_s * tanhf(c);
  c_st[idx]     = c;                                   // keep cell state f32
  h_bf[idx]     = (bf16_t)h;                           // next-step GEMM input
  out_bf_t[idx] = (bf16_t)h;                           // projection A matrix
}

// --- projection: logits[r][v] = outputs[r] . W_out[v] + b_out[v] -----------
// M=2048, N=32000, K=512.  LDS-tiled: each 256-thread WG computes a 128x128
// output block; per 32-wide K-chunk an 8KB A-slab + 8KB B-slab are staged in
// LDS and every wave runs 2Mx4N WMMAs from ds_load fragments (8x operand
// reuse each way -> ~1 GB total L2 traffic instead of ~5 GB).
__global__ __launch_bounds__(256)
void k_proj(const bf16_t* __restrict__ A,              // [T_*B_][H_]
            const bf16_t* __restrict__ Wo,             // [V_][H_]
            const float* __restrict__ b_out,
            float* __restrict__ out) {                 // [B_][T_][V_]
  __shared__ __align__(64) bf16_t Asub[128][32];       // 8 KB
  __shared__ __align__(64) bf16_t Bsub[128][32];       // 8 KB
  const int tid = threadIdx.x;
  const int mgroup = blockIdx.x & 15;                  // 2048/128
  const int ngroup = blockIdx.x >> 4;                  // 0..249 (32000/128)
  const int mbase = mgroup * 128;
  const int nbase = ngroup * 128;
  const int wv = tid >> 5, lane = tid & 31;
  const int m = lane & 15, half = lane >> 4;
  const int lmt0 = (wv & 3) * 2;                       // wave's 2 local M-tiles
  const int lnt0 = (wv >> 2) * 4;                      // wave's 4 local N-tiles

  v8f acc[2][4] = {};

  for (int kb = 0; kb < H_; kb += 32) {
    // cooperative stage: 512 chunks of 8 bf16 for each slab (2 per thread)
#pragma unroll
    for (int c = 0; c < 2; ++c) {
      const int ch = tid * 2 + c;                      // 0..511
      const int r = ch >> 2, col = (ch & 3) * 8;
      *reinterpret_cast<v8bf*>(&Asub[r][col]) =
          *reinterpret_cast<const v8bf*>(A + (size_t)(mbase + r) * H_ + kb + col);
      *reinterpret_cast<v8bf*>(&Bsub[r][col]) =
          *reinterpret_cast<const v8bf*>(Wo + (size_t)(nbase + r) * H_ + kb + col);
    }
    if (kb + 32 < H_)
      __builtin_prefetch(Wo + (size_t)(nbase + tid) * H_ + kb + 32, 0, 0);
    __syncthreads();

    v16bf af[2], bfrag[4];
#pragma unroll
    for (int i = 0; i < 2; ++i) {
      const bf16_t* ar = &Asub[(lmt0 + i) * 16 + m][0];
      const v8bf lo = *reinterpret_cast<const v8bf*>(ar + half * 16);
      const v8bf hi = *reinterpret_cast<const v8bf*>(ar + half * 16 + 8);
#pragma unroll
      for (int e = 0; e < 8; ++e) { af[i][e] = lo[e]; af[i][e + 8] = hi[e]; }
    }
#pragma unroll
    for (int j = 0; j < 4; ++j)
      bfrag[j] = *reinterpret_cast<const v16bf*>(&Bsub[(lnt0 + j) * 16 + m][half * 16]);

#pragma unroll
    for (int i = 0; i < 2; ++i)
#pragma unroll
      for (int j = 0; j < 4; ++j)
        acc[i][j] = __builtin_amdgcn_wmma_f32_16x16x32_bf16(
            false, af[i], false, bfrag[j], (short)0, acc[i][j], false, false);
    __syncthreads();
  }

  // epilogue: bias add + scatter into [B][T][V]
#pragma unroll
  for (int i = 0; i < 2; ++i) {
#pragma unroll
    for (int j = 0; j < 4; ++j) {
      const int n = nbase + (lnt0 + j) * 16 + m;
      const float bias = b_out[n];
#pragma unroll
      for (int r = 0; r < 8; ++r) {
        const int rm = mbase + (lmt0 + i) * 16 + half * 8 + r;  // row = t*B_ + b
        const int tt = rm >> 5, bb = rm & 31;
        out[((size_t)bb * T_ + tt) * V_ + n] = acc[i][j][r] + bias;
      }
    }
  }
}

// --- log_softmax in place over last dim of [B_][T_][V_] --------------------
__global__ __launch_bounds__(256)
void k_log_softmax(float* __restrict__ out) {
  __shared__ float red[256];
  float* p = out + (size_t)blockIdx.x * V_;
  const int tid = threadIdx.x;
  float mx = -INFINITY;
  for (int v = tid; v < V_; v += 256) mx = fmaxf(mx, p[v]);
  red[tid] = mx; __syncthreads();
  for (int s = 128; s > 0; s >>= 1) {
    if (tid < s) red[tid] = fmaxf(red[tid], red[tid + s]);
    __syncthreads();
  }
  const float rowmax = red[0]; __syncthreads();
  float sum = 0.f;
  for (int v = tid; v < V_; v += 256) sum += __expf(p[v] - rowmax);
  red[tid] = sum; __syncthreads();
  for (int s = 128; s > 0; s >>= 1) {
    if (tid < s) red[tid] += red[tid + s];
    __syncthreads();
  }
  const float lse = rowmax + __logf(red[0]);
  for (int v = tid; v < V_; v += 256) p[v] = p[v] - lse;
}

// ---------------------------------------------------------------------------
extern "C" void kernel_launch(void* const* d_in, const int* in_sizes, int n_in,
                              void* d_out, int out_size, void* d_ws, size_t ws_size,
                              hipStream_t stream) {
  (void)in_sizes; (void)n_in; (void)out_size; (void)ws_size;
  const float*     h0    = (const float*)d_in[1];
  const float*     c0    = (const float*)d_in[2];
  const float*     emb   = (const float*)d_in[3];
  const float*     W_ih  = (const float*)d_in[4];
  const float*     W_hh  = (const float*)d_in[5];
  const float*     b_ih  = (const float*)d_in[6];
  const float*     b_hh  = (const float*)d_in[7];
  const float*     W_out = (const float*)d_in[8];
  const float*     b_out = (const float*)d_in[9];
  const long long* dst   = (const long long*)d_in[10];
  float* out = (float*)d_out;

  // workspace carve-up (256B aligned)
  char* ws = (char*)d_ws;
  size_t off = 0;
  auto carve = [&](size_t bytes) {
    void* p = ws + off;
    off = (off + bytes + 255) & ~(size_t)255;
    return p;
  };
  bf16_t* Wcat   = (bf16_t*)carve((size_t)G_ * KC_ * sizeof(bf16_t)); //  4 MB
  bf16_t* Wo     = (bf16_t*)carve((size_t)V_ * H_ * sizeof(bf16_t));  // 32 MB
  bf16_t* x_bf   = (bf16_t*)carve((size_t)T_ * B_ * D_ * sizeof(bf16_t));
  bf16_t* out_bf = (bf16_t*)carve((size_t)T_ * B_ * H_ * sizeof(bf16_t));
  float*  gates  = (float*) carve((size_t)B_ * G_ * sizeof(float));
  float*  c_st   = (float*) carve((size_t)B_ * H_ * sizeof(float));
  bf16_t* h_bf   = (bf16_t*)carve((size_t)B_ * H_ * sizeof(bf16_t));

  // 1) one-time conversions / packing
  k_pack_wcat   <<<(G_ * KC_) / 256, 256, 0, stream>>>(W_ih, W_hh, Wcat);
  k_convert_wout<<<((size_t)V_ * H_) / 256, 256, 0, stream>>>(W_out, Wo);
  k_embed       <<<(T_ * B_ * D_) / 256, 256, 0, stream>>>(emb, dst, x_bf);
  k_init_state  <<<(B_ * H_) / 256, 256, 0, stream>>>(h0, c0, h_bf, c_st);

  // 2) sequential LSTM scan (stream order provides the h/c dependency)
  for (int t = 0; t < T_; ++t) {
    k_lstm_gates <<<32, 256, 0, stream>>>(x_bf + (size_t)t * B_ * D_, h_bf,
                                          Wcat, b_ih, b_hh, gates);
    k_lstm_update<<<(B_ * H_) / 256, 256, 0, stream>>>(gates, c_st, h_bf,
                                          out_bf + (size_t)t * B_ * H_);
  }

  // 3) vocab projection (dominant GEMM) straight into d_out, then log_softmax
  k_proj       <<<16 * 250, 256, 0, stream>>>(out_bf, Wo, b_out, out);
  k_log_softmax<<<B_ * T_, 256, 0, stream>>>(out);
}
